// GNN_Model_62526133895289
// MI455X (gfx1250) — matrix operand order repaired and checked by reading, not compiled, not run on hardware
//
#include <hip/hip_runtime.h>
#include <hip/hip_bf16.h>

typedef __attribute__((ext_vector_type(16))) _Float16 v16h;
typedef __attribute__((ext_vector_type(8)))  float    v8f;

#define N_NODES  50000
#define F_IN     16
#define N_HIDDEN 32
#define N_EDGES  1000000
#define EDGE_S   8
#define BN_EPS   0.001f

// GEMM view: msgs[1M x 32] = (e' (x) x_src)[1M x 144, padded 160] @ W'[160 x 32]
#define K_PAD    160
#define N_KCHUNK 5       // 160 / 32
#define N_NTILE  2       // 32 / 16
#define EPB      128     // edges per block (8 waves x 16 rows)
#define TPB      256

// workspace layout (floats unless noted)
#define POOL_OFF      (N_NODES * N_HIDDEN)          // 1,600,000
#define ZERO_N        (N_NODES * N_HIDDEN + 32)     // agg + pooled
#define BPK_OFF_BYTES ((size_t)ZERO_N * 4)          // 6,400,128 (32B aligned)

// ---------------------------------------------------------------- zero scratch
__global__ void zero_ws(float* __restrict__ ws, int n) {
    int i = blockIdx.x * blockDim.x + threadIdx.x;
    if (i < n) ws[i] = 0.f;
}

// ---------------------------------------------------- pack B = [Wk ; bk ; 0pad]
// Layout matches v_wmma_f32_16x16x32_f16 B operand (32x16 f16, 8 VGPRs):
//   lanes 0-15  : K = (v>=4?16:0) + 2(v&3), +1      (N = lane)
//   lanes 16-31 : K = (v>=4?16:0) + 8 + 2(v&3), +1  (N = lane-16)
// Stored as Bpk[((t*5 + c)*32 + lane)*8 + v] packed 2xf16 per dword.
__global__ void prep_B(const float* __restrict__ Wk, const float* __restrict__ bk,
                       unsigned int* __restrict__ Bpk) {
    int i = blockIdx.x * blockDim.x + threadIdx.x;
    if (i >= N_NTILE * N_KCHUNK * 32 * 8) return;
    int v    = i & 7;
    int lane = (i >> 3) & 31;
    int rest = i >> 8;            // t*5 + c
    int c    = rest % N_KCHUNK;
    int t    = rest / N_KCHUNK;
    int half = lane >> 4;
    int n    = (lane & 15) + 16 * t;
    int k0   = 32 * c + (v >= 4 ? 16 : 0) + half * 8 + 2 * (v & 3);

    float f[2];
    #pragma unroll
    for (int b = 0; b < 2; ++b) {
        int k = k0 + b;
        int s = k >> 4, ff = k & 15;
        float val = 0.f;
        if (s < 8)       val = Wk[s * (F_IN * N_HIDDEN) + ff * N_HIDDEN + n];
        else if (s == 8) val = bk[ff * N_HIDDEN + n];
        f[b] = val;
    }
    union { _Float16 h[2]; unsigned int u; } p;
    p.h[0] = (_Float16)f[0];
    p.h[1] = (_Float16)f[1];
    Bpk[i] = p.u;
}

// ------------------------------------------- fused edge GEMM (WMMA) + scatter
__global__ void __launch_bounds__(TPB)
edge_gemm_scatter(const float* __restrict__ x, const int* __restrict__ ei,
                  const float* __restrict__ e, const unsigned int* __restrict__ Bpk,
                  float* __restrict__ agg) {
    __shared__ float sE[EPB][10];   // e'[0..7]=e, [8]=valid?1:0, [9]=0 (K pad)
    __shared__ float sX[EPB][16];   // gathered x[src]
    __shared__ int   sT[EPB];
    __shared__ int   sS[EPB];

    const int  tid  = threadIdx.x;
    const long base = (long)blockIdx.x * EPB;

    // phase 1: indices + e-features (no dependence on sS yet)
    for (int i = tid; i < EPB; i += TPB) {
        long eg = base + i;
        int srci = 0, tgti = -1; float one = 0.f;
        if (eg < N_EDGES) { srci = ei[2 * eg]; tgti = ei[2 * eg + 1]; one = 1.f; }
        sS[i] = srci; sT[i] = tgti;
        sE[i][8] = one; sE[i][9] = 0.f;
    }
    for (int i = tid; i < EPB * EDGE_S; i += TPB) {
        int r = i >> 3, s = i & 7;
        long eg = base + r;
        sE[r][s] = (eg < N_EDGES) ? e[eg * EDGE_S + s] : 0.f;
    }
    __syncthreads();
    // phase 2: gather x rows (x is L2-resident: 3.2 MB)
    for (int i = tid; i < EPB * F_IN; i += TPB) {
        int r = i >> 4, f = i & 15;
        long eg = base + r;
        sX[r][f] = (eg < N_EDGES) ? x[(long)sS[r] * F_IN + f] : 0.f;
    }
    __syncthreads();

    const int lane = tid & 31;
    const int wave = tid >> 5;
    const int half = lane >> 4;
    const int row  = wave * 16 + (lane & 15);

    // per-lane x slice for the f16 A layout: f = half*8 + j (constant across chunks)
    _Float16 xh[8];
    #pragma unroll
    for (int j = 0; j < 8; ++j) xh[j] = (_Float16)sX[row][half * 8 + j];

    v8f acc0 = {}; v8f acc1 = {};
    const v16h* Bp = (const v16h*)Bpk;

    #pragma unroll
    for (int c = 0; c < N_KCHUNK; ++c) {
        // A element (VGPR v, sub b): k = 32c + (v>=4?16:0) + half*8 + 2*(v&3) + b
        //   => s = 2c + (v>=4), f = half*8 + 2*(v&3) + b
        _Float16 e0 = (_Float16)sE[row][2 * c];
        _Float16 e1 = (_Float16)sE[row][2 * c + 1];
        v16h a;
        #pragma unroll
        for (int v = 0; v < 8; ++v) {
            _Float16 es = (v < 4) ? e0 : e1;
            a[2 * v]     = es * xh[2 * (v & 3)];
            a[2 * v + 1] = es * xh[2 * (v & 3) + 1];
        }
        v16h b0 = Bp[(0 * N_KCHUNK + c) * 32 + lane];
        v16h b1 = Bp[(1 * N_KCHUNK + c) * 32 + lane];
        acc0 = __builtin_amdgcn_wmma_f32_16x16x32_f16(false, a, false, b0,
                                                      (short)0, acc0, false, false);
        acc1 = __builtin_amdgcn_wmma_f32_16x16x32_f16(false, a, false, b1,
                                                      (short)0, acc1, false, false);
    }

    // scatter D: VGPR j -> M = j + half*8, N = (lane&15) + 16*tile
    #pragma unroll
    for (int j = 0; j < 8; ++j) {
        int m   = j + half * 8;
        int tgt = sT[wave * 16 + m];
        if (tgt >= 0) {
            int n = lane & 15;
            atomicAdd(&agg[(long)tgt * N_HIDDEN + n],      acc0[j]);
            atomicAdd(&agg[(long)tgt * N_HIDDEN + n + 16], acc1[j]);
        }
    }
}

// ------------------------------------- node: root GEMM + ReLU + BN + pool-sum
__global__ void __launch_bounds__(256)
node_bn_pool(const float* __restrict__ x, const float* __restrict__ agg,
             const float* __restrict__ Wr, const float* __restrict__ bconv,
             const float* __restrict__ gamma, const float* __restrict__ beta,
             const float* __restrict__ mean, const float* __restrict__ var,
             float* __restrict__ pooled) {
    __shared__ float sW[F_IN * N_HIDDEN];
    __shared__ float sScale[N_HIDDEN], sShift[N_HIDDEN], sB[N_HIDDEN], sP[N_HIDDEN];
    const int tid = threadIdx.x;

    for (int i = tid; i < F_IN * N_HIDDEN; i += 256) sW[i] = Wr[i];
    if (tid < N_HIDDEN) {
        float sc   = gamma[tid] * rsqrtf(var[tid] + BN_EPS);
        sScale[tid] = sc;
        sShift[tid] = beta[tid] - mean[tid] * sc;
        sB[tid]     = bconv[tid];
        sP[tid]     = 0.f;
    }
    __syncthreads();

    int n = blockIdx.x * 256 + tid;
    if (n < N_NODES) {
        float xv[F_IN];
        #pragma unroll
        for (int f = 0; f < F_IN; ++f) xv[f] = x[(long)n * F_IN + f];
        #pragma unroll
        for (int oo = 0; oo < N_HIDDEN; ++oo) {
            int o = (tid + oo) & (N_HIDDEN - 1);   // rotate -> conflict-free ds_add
            float s = agg[(long)n * N_HIDDEN + o] + sB[o];
            #pragma unroll
            for (int f = 0; f < F_IN; ++f) s += xv[f] * sW[f * N_HIDDEN + o];
            float h  = fmaxf(s, 0.f);
            float hn = h * sScale[o] + sShift[o];
            atomicAdd(&sP[o], hn);
        }
    }
    __syncthreads();
    if (tid < N_HIDDEN) atomicAdd(&pooled[tid], sP[tid]);
}

// ------------------------------------------------------- final 1x32 @ 32x3
__global__ void final_dense(const float* __restrict__ pooled,
                            const float* __restrict__ Wd,
                            const float* __restrict__ bd,
                            float* __restrict__ out) {
    int j = threadIdx.x;
    if (j < 3) {
        float s = bd[j];
        #pragma unroll
        for (int o = 0; o < N_HIDDEN; ++o) s += pooled[o] * Wd[o * 3 + j];
        out[j] = s;
    }
}

// ---------------------------------------------------------------------- launch
extern "C" void kernel_launch(void* const* d_in, const int* in_sizes, int n_in,
                              void* d_out, int out_size, void* d_ws, size_t ws_size,
                              hipStream_t stream) {
    const float* x  = (const float*)d_in[0];
    const int*   ei = (const int*)  d_in[1];
    const float* e  = (const float*)d_in[2];
    const float* Wk = (const float*)d_in[3];
    const float* bk = (const float*)d_in[4];
    const float* Wr = (const float*)d_in[5];
    const float* bc = (const float*)d_in[6];
    const float* ga = (const float*)d_in[7];
    const float* be = (const float*)d_in[8];
    const float* mm = (const float*)d_in[9];
    const float* mv = (const float*)d_in[10];
    const float* Wd = (const float*)d_in[11];
    const float* bd = (const float*)d_in[12];
    float* out = (float*)d_out;

    // workspace: [agg 50000x32 f32][pooled 32 f32][Bpk 2560 u32]  (~6.41 MB)
    float*        ws     = (float*)d_ws;
    float*        agg    = ws;
    float*        pooled = ws + POOL_OFF;
    unsigned int* Bpk    = (unsigned int*)((char*)d_ws + BPK_OFF_BYTES);

    zero_ws<<<(ZERO_N + 255) / 256, 256, 0, stream>>>(ws, ZERO_N);
    prep_B<<<(N_NTILE * N_KCHUNK * 32 * 8 + 255) / 256, 256, 0, stream>>>(Wk, bk, Bpk);

    int nblk = (N_EDGES + EPB - 1) / EPB;
    edge_gemm_scatter<<<nblk, TPB, 0, stream>>>(x, ei, e, Bpk, agg);

    node_bn_pool<<<(N_NODES + 255) / 256, 256, 0, stream>>>(x, agg, Wr, bc, ga, be, mm, mv, pooled);
    final_dense<<<1, 32, 0, stream>>>(pooled, Wd, bd, out);
}
